// Attention_82248623718459
// MI455X (gfx1250) — matrix operand order repaired and checked by reading, not compiled, not run on hardware
//
#include <hip/hip_runtime.h>
#include <hip/hip_bf16.h>

#define S_LEN 2048
#define H_DIM 1024
#define NH_   16
#define D_    64
#define H3_   3072

typedef __attribute__((ext_vector_type(16))) _Float16 v16h;
typedef __attribute__((ext_vector_type(8)))  _Float16 v8h;
typedef __attribute__((ext_vector_type(8)))  float    v8f;
typedef __attribute__((ext_vector_type(4)))  unsigned int u32x4;
typedef __attribute__((ext_vector_type(8)))  int          i32x8;
typedef __attribute__((ext_vector_type(4)))  int          i32x4;

static __device__ __forceinline__ v8f wmma_f16(v16h a, v16h b, v8f c) {
  return __builtin_amdgcn_wmma_f32_16x16x32_f16(false, a, false, b, (short)0, c,
                                                false, false);
}

static __device__ __forceinline__ v16h cat8(v8h lo, v8h hi) {
  v16h r;
#pragma unroll
  for (int j = 0; j < 8; ++j) { r[j] = lo[j]; r[j + 8] = hi[j]; }
  return r;
}

#if __has_builtin(__builtin_amdgcn_tensor_load_to_lds)
#define HAVE_TDM 1
#else
#define HAVE_TDM 0
#endif

// ---- Tensor Data Mover descriptor helpers (CDNA5 ISA ch.8) -----------------
// Group1 (loop-invariant): data_size=2B, tensor dims == tile dims, dim0 stride.
static __device__ __forceinline__ i32x8 tdm_group1_2d(unsigned tile_d0,
                                                      unsigned tile_d1,
                                                      unsigned stride_elems) {
  i32x8 g1;
  g1[0] = (int)(1u << 16);                                    // data_size=2B
  g1[1] = (int)((tile_d0 & 0xFFFFu) << 16);                   // tensor_dim0 lo
  g1[2] = (int)(((tile_d0 >> 16) & 0xFFFFu) |
                ((tile_d1 & 0xFFFFu) << 16));                 // d0 hi | d1 lo
  g1[3] = (int)(((tile_d1 >> 16) & 0xFFFFu) | (tile_d0 << 16));  // d1 hi | tile_d0
  g1[4] = (int)(tile_d1 & 0xFFFFu);                           // tile_d1, tile_d2=0
  g1[5] = (int)stride_elems;                                  // dim0 stride lo
  g1[6] = 0;
  g1[7] = 0;
  return g1;
}

// Group0 (per-issue): count=1, LDS byte addr, 57-bit global addr, type=2.
static __device__ __forceinline__ u32x4 tdm_group0(unsigned lds_addr,
                                                   unsigned long long ga) {
  u32x4 g0;
  g0[0] = 1u;
  g0[1] = lds_addr;
  g0[2] = (unsigned)ga;
  g0[3] = (unsigned)((ga >> 32) & 0x01FFFFFFu) | (2u << 30);
  return g0;
}

static __device__ __forceinline__ void tdm_issue(u32x4 g0, i32x8 g1) {
#if HAVE_TDM
  i32x4 z4 = {0, 0, 0, 0};
#if defined(__clang_major__) && (__clang_major__ >= 23)
  i32x8 z8 = {0, 0, 0, 0, 0, 0, 0, 0};
  __builtin_amdgcn_tensor_load_to_lds(g0, g1, z4, z4, z8, 0);
#else
  __builtin_amdgcn_tensor_load_to_lds(g0, g1, z4, z4, 0);
#endif
#endif
}

// ---------------------------------------------------------------------------
// fp32 -> fp16 elementwise convert (x, E)
// ---------------------------------------------------------------------------
__global__ void cvt_f16_kernel(const float* __restrict__ src,
                               _Float16* __restrict__ dst, int n) {
  int i = blockIdx.x * blockDim.x + threadIdx.x;
  if (i < n) dst[i] = (_Float16)src[i];
}

// ---------------------------------------------------------------------------
// fp32 [K,N] -> fp16 [N,K] tiled transpose (weights): all GEMM B-fragments
// become contiguous 32B vector loads.
// ---------------------------------------------------------------------------
__global__ __launch_bounds__(256) void transpose_cvt_kernel(
    const float* __restrict__ W, _Float16* __restrict__ Wt, int K, int N) {
  __shared__ _Float16 tile[32][33];
  const int nb = blockIdx.x * 32;
  const int kb = blockIdx.y * 32;
  const int tx = threadIdx.x;       // 0..31
  const int ty = threadIdx.y;       // 0..7
  for (int r = ty; r < 32; r += 8)
    tile[r][tx] = (_Float16)W[(size_t)(kb + r) * N + nb + tx];
  __syncthreads();
  for (int r = ty; r < 32; r += 8)
    Wt[(size_t)(nb + r) * K + kb + tx] = tile[tx][r];
}

// ---------------------------------------------------------------------------
// QKV GEMM: xh[2048,1024]f16 @ Wt^T (Wt = [3072,1024]f16) + bias.
// One wave -> 32(M) x 64(N) tile; 8 WMMA per k-step with A/B reuse.
// 64-wide N tiles align with q/k/v and head boundaries -> uniform routing.
// ---------------------------------------------------------------------------
__global__ __launch_bounds__(32) void qkv_gemm_kernel(
    const _Float16* __restrict__ xh, const _Float16* __restrict__ Wt,
    const float* __restrict__ bias, float* __restrict__ present,
    _Float16* __restrict__ qh, _Float16* __restrict__ kh,
    _Float16* __restrict__ vt) {
  const int n0 = blockIdx.x * 64;
  const int m0 = blockIdx.y * 32;
  const int lane = threadIdx.x;
  const int lm = lane & 15, hl = lane >> 4;

  v8f acc[2][4] = {};
  for (int k0 = 0; k0 < H_DIM; k0 += 32) {
    v16h a[2];
#pragma unroll
    for (int mt = 0; mt < 2; ++mt) {
      const _Float16* ar = xh + (size_t)(m0 + mt * 16 + lm) * H_DIM + k0;
      a[mt] = cat8(*(const v8h*)(ar + 16 * hl), *(const v8h*)(ar + 8 + 16 * hl));
    }
#pragma unroll
    for (int nt = 0; nt < 4; ++nt) {
      const v16h b =
          *(const v16h*)(Wt + (size_t)(n0 + nt * 16 + lm) * H_DIM + k0 + 16 * hl);
      acc[0][nt] = wmma_f16(a[0], b, acc[0][nt]);
      acc[1][nt] = wmma_f16(a[1], b, acc[1][nt]);
    }
  }

  const int which = n0 >> 10;             // uniform: 0=q 1=k 2=v
  const int head  = (n0 & 1023) >> 6;     // uniform
#pragma unroll
  for (int nt = 0; nt < 4; ++nt) {
    const int d = nt * 16 + lm;
    const float bv = bias[n0 + nt * 16 + lm];
#pragma unroll
    for (int mt = 0; mt < 2; ++mt) {
#pragma unroll
      for (int r = 0; r < 8; ++r) {
        const int srow = m0 + mt * 16 + r + 8 * hl;
        const float val = acc[mt][nt][r] + bv;
        const size_t off = ((size_t)head * S_LEN + srow) * D_ + d;
        if (which == 0) {
          qh[off] = (_Float16)val;
        } else if (which == 1) {
          kh[off] = (_Float16)val;
          present[off] = val;                                    // present[:,0]=k
        } else {
          vt[((size_t)head * D_ + d) * S_LEN + srow] = (_Float16)val;
          present[(size_t)NH_ * S_LEN * D_ + off] = val;         // present[:,1]=v
        }
      }
    }
  }
}

// ---------------------------------------------------------------------------
// Flash attention with relative positions (skew identity:
// rel[i,j] = q[i] . E[S-1-i+j], valid for j<=i).
// One wave = (head, 16 query rows).  K/V tiles (32 keys) are double-buffered
// into LDS by the Tensor Data Mover; DMA of block j+1 overlaps WMMA of j.
// ---------------------------------------------------------------------------
__global__ __launch_bounds__(32) void flash_attn_rel_kernel(
    const _Float16* __restrict__ qh, const _Float16* __restrict__ kh,
    const _Float16* __restrict__ vt, const _Float16* __restrict__ Eh,
    _Float16* __restrict__ aoh) {
  const int h  = blockIdx.x;
  const int i0 = blockIdx.y * 16;
  const int lane = threadIdx.x;
  const int lm = lane & 15, hl = lane >> 4;

  __shared__ __align__(32) _Float16 sK[2][32][64];  // key tile  [key][d]
  __shared__ __align__(32) _Float16 sV[2][64][32];  // value^T   [d][key]
  __shared__ float    sS[16][32];
  __shared__ float    sG[16][48];
  __shared__ __align__(16) _Float16 sP[16][32];
  __shared__ float    sAlpha[16];
  __shared__ float    sLinv[16];

  const _Float16* Q  = qh + (size_t)h * S_LEN * D_;
  const _Float16* K  = kh + (size_t)h * S_LEN * D_;
  const _Float16* Vt = vt + (size_t)h * D_ * S_LEN;
  const _Float16* E  = Eh + (size_t)h * S_LEN * D_;

  v16h aq0, aq1;
  {
    const _Float16* qr = Q + (size_t)(i0 + lm) * D_;
    aq0 = cat8(*(const v8h*)(qr + 16 * hl), *(const v8h*)(qr + 8 + 16 * hl));
    aq1 = cat8(*(const v8h*)(qr + 32 + 16 * hl), *(const v8h*)(qr + 40 + 16 * hl));
  }

  v8f acc[4] = {};
  float mrow = -3.0e38f, lrow = 0.0f;
  const int nblocks = i0 / 32 + 1;   // causal: keys 0 .. i0+15

#if HAVE_TDM
  // Loop-invariant descriptor parts hoisted; only addresses change per block.
  const i32x8 g1K = tdm_group1_2d(64, 32, 64);      // 32 keys x 64 dims, rows contig
  const i32x8 g1V = tdm_group1_2d(32, 64, S_LEN);   // 64 dims x 32 keys, stride S
  const unsigned ldsK0 = (unsigned)(unsigned long long)&sK[0][0][0];
  const unsigned ldsK1 = (unsigned)(unsigned long long)&sK[1][0][0];
  const unsigned ldsV0 = (unsigned)(unsigned long long)&sV[0][0][0];
  const unsigned ldsV1 = (unsigned)(unsigned long long)&sV[1][0][0];
  const unsigned long long gaK = (unsigned long long)K;
  const unsigned long long gaV = (unsigned long long)Vt;
  tdm_issue(tdm_group0(ldsK0, gaK), g1K);
  tdm_issue(tdm_group0(ldsV0, gaV), g1V);
#endif

  for (int blk = 0; blk < nblocks; ++blk) {
    const int j0  = blk * 32;
    const int cur = blk & 1;

#if HAVE_TDM
    if (blk + 1 < nblocks) {
      asm volatile("s_wait_dscnt 0" ::: "memory");   // LDS reads done before DMA overwrite
      tdm_issue(tdm_group0(cur ? ldsK0 : ldsK1,
                           gaK + (unsigned long long)(j0 + 32) * D_ * 2), g1K);
      tdm_issue(tdm_group0(cur ? ldsV0 : ldsV1,
                           gaV + (unsigned long long)(j0 + 32) * 2), g1V);
      __builtin_amdgcn_s_wait_tensorcnt(2);          // current tiles landed
    } else {
      __builtin_amdgcn_s_wait_tensorcnt(0);
    }
    asm volatile("" ::: "memory");  // pin LDS reads behind the tensorcnt wait
#else
    {  // cooperative fallback staging
#pragma unroll
      for (int t = 0; t < 8; ++t) {
        const int c = t * 32 + lane;                 // 0..255 chunks of 8 halves
        ((v8h*)&sK[cur][0][0])[c] = *(const v8h*)(K + (size_t)j0 * D_ + c * 8);
        const int row = c >> 2, c8 = c & 3;
        ((v8h*)&sV[cur][0][0])[c] =
            *(const v8h*)(Vt + (size_t)row * S_LEN + j0 + c8 * 8);
      }
      __syncthreads();
    }
#endif

    // ---- scores: Q(16x64) @ K_tile^T(64x32), B-frags from LDS -------------
#pragma unroll
    for (int nt = 0; nt < 2; ++nt) {
      const v16h bk0 = *(const v16h*)&sK[cur][nt * 16 + lm][16 * hl];
      const v16h bk1 = *(const v16h*)&sK[cur][nt * 16 + lm][32 + 16 * hl];
      v8f cS = {};
      cS = wmma_f16(aq0, bk0, cS);
      cS = wmma_f16(aq1, bk1, cS);
#pragma unroll
      for (int r = 0; r < 8; ++r) sS[r + 8 * hl][nt * 16 + lm] = cS[r];
    }

    // ---- relative band: G = Q(16x64) @ Eband^T(64x48) ---------------------
    const int base = S_LEN - 16 - i0 + j0;
#pragma unroll
    for (int nt = 0; nt < 3; ++nt) {
      int er = base + nt * 16 + lm;
      er = er < 0 ? 0 : (er > S_LEN - 1 ? S_LEN - 1 : er);  // OOB rows masked later
      const _Float16* erow = E + (size_t)er * D_;
      const v16h be0 = *(const v16h*)(erow + 16 * hl);
      const v16h be1 = *(const v16h*)(erow + 32 + 16 * hl);
      v8f cG = {};
      cG = wmma_f16(aq0, be0, cG);
      cG = wmma_f16(aq1, be1, cG);
#pragma unroll
      for (int r = 0; r < 8; ++r) sG[r + 8 * hl][nt * 16 + lm] = cG[r];
    }
    __syncthreads();

    // ---- online softmax (lanes 0..15, one row each) -----------------------
    if (lane < 16) {
      const int row = lane;
      const int gi  = i0 + row;
      float sc[32];
      float tmax = -3.0e38f;
#pragma unroll
      for (int dj = 0; dj < 32; ++dj) {
        float w = (sS[row][dj] + sG[row][15 + dj - row]) * 0.125f;
        if (j0 + dj > gi) w = -1.0e30f;   // causal mask
        sc[dj] = w;
        tmax = fmaxf(tmax, w);
      }
      const float mnew  = fmaxf(mrow, tmax);
      const float alpha = __expf(mrow - mnew);
      float psum = 0.0f;
#pragma unroll
      for (int dj = 0; dj < 32; ++dj) {
        const float p = __expf(sc[dj] - mnew);
        psum += p;
        sP[row][dj] = (_Float16)p;
      }
      lrow = lrow * alpha + psum;
      mrow = mnew;
      sAlpha[row] = alpha;
    }
    __syncthreads();

    // ---- rescale accumulator + P(16x32) @ V_tile(32x64) -------------------
    const v16h ap = cat8(*(const v8h*)&sP[lm][16 * hl],
                         *(const v8h*)&sP[lm][8 + 16 * hl]);
    float fac[8];
#pragma unroll
    for (int r = 0; r < 8; ++r) fac[r] = sAlpha[r + 8 * hl];
#pragma unroll
    for (int nt = 0; nt < 4; ++nt) {
#pragma unroll
      for (int r = 0; r < 8; ++r) acc[nt][r] *= fac[r];
      const v16h bv = *(const v16h*)&sV[cur][nt * 16 + lm][16 * hl];
      acc[nt] = wmma_f16(ap, bv, acc[nt]);
    }
    __syncthreads();   // protect sS/sG/sP for next block
  }

  if (lane < 16) sLinv[lane] = 1.0f / lrow;
  __syncthreads();

#pragma unroll
  for (int nt = 0; nt < 4; ++nt) {
#pragma unroll
    for (int r = 0; r < 8; ++r) {
      aoh[(size_t)(i0 + r + 8 * hl) * H_DIM + h * D_ + nt * 16 + lm] =
          (_Float16)(acc[nt][r] * sLinv[r + 8 * hl]);
    }
  }
}

// ---------------------------------------------------------------------------
// Output projection: aoh[2048,1024]f16 @ Wt^T + bias -> d_out f32
// ---------------------------------------------------------------------------
__global__ __launch_bounds__(32) void proj_gemm_kernel(
    const _Float16* __restrict__ aoh, const _Float16* __restrict__ Wt,
    const float* __restrict__ bias, float* __restrict__ out) {
  const int n0 = blockIdx.x * 64;
  const int m0 = blockIdx.y * 32;
  const int lane = threadIdx.x;
  const int lm = lane & 15, hl = lane >> 4;

  v8f acc[2][4] = {};
  for (int k0 = 0; k0 < H_DIM; k0 += 32) {
    v16h a[2];
#pragma unroll
    for (int mt = 0; mt < 2; ++mt) {
      const _Float16* ar = aoh + (size_t)(m0 + mt * 16 + lm) * H_DIM + k0;
      a[mt] = cat8(*(const v8h*)(ar + 16 * hl), *(const v8h*)(ar + 8 + 16 * hl));
    }
#pragma unroll
    for (int nt = 0; nt < 4; ++nt) {
      const v16h b =
          *(const v16h*)(Wt + (size_t)(n0 + nt * 16 + lm) * H_DIM + k0 + 16 * hl);
      acc[0][nt] = wmma_f16(a[0], b, acc[0][nt]);
      acc[1][nt] = wmma_f16(a[1], b, acc[1][nt]);
    }
  }
#pragma unroll
  for (int nt = 0; nt < 4; ++nt) {
    const float bv = bias[n0 + nt * 16 + lm];
#pragma unroll
    for (int mt = 0; mt < 2; ++mt) {
#pragma unroll
      for (int r = 0; r < 8; ++r) {
        out[(size_t)(m0 + mt * 16 + r + 8 * hl) * H_DIM + n0 + nt * 16 + lm] =
            acc[mt][nt][r] + bv;
      }
    }
  }
}

// ---------------------------------------------------------------------------
extern "C" void kernel_launch(void* const* d_in, const int* in_sizes, int n_in,
                              void* d_out, int out_size, void* d_ws,
                              size_t ws_size, hipStream_t stream) {
  (void)in_sizes; (void)n_in; (void)out_size; (void)ws_size;
  const float* x      = (const float*)d_in[0];
  // d_in[1] causal mask: implemented analytically.
  const float* W_attn = (const float*)d_in[2];
  const float* b_attn = (const float*)d_in[3];
  const float* W_proj = (const float*)d_in[4];
  const float* b_proj = (const float*)d_in[5];
  const float* E      = (const float*)d_in[6];

  float* out     = (float*)d_out;
  float* present = out + (size_t)S_LEN * H_DIM;   // [2][NH][S][D] f32

  char* ws = (char*)d_ws;
  const size_t MB4 = (size_t)S_LEN * H_DIM * sizeof(_Float16);  // 4 MB units
  _Float16* xh      = (_Float16*)(ws);                  // [S][H]      4 MB
  _Float16* qh      = (_Float16*)(ws + 1 * MB4);        // [NH][S][D]  4 MB
  _Float16* kh      = (_Float16*)(ws + 2 * MB4);        // [NH][S][D]  4 MB
  _Float16* vt      = (_Float16*)(ws + 3 * MB4);        // [NH][D][S]  4 MB
  _Float16* Eh      = (_Float16*)(ws + 4 * MB4);        // [NH][S][D]  4 MB
  _Float16* aoh     = (_Float16*)(ws + 5 * MB4);        // [S][H]      4 MB
  _Float16* Wt_attn = (_Float16*)(ws + 6 * MB4);        // [3072][1024] 6 MB
  _Float16* Wt_proj = (_Float16*)(ws + 6 * MB4 + (size_t)H3_ * H_DIM * 2); // 2 MB

  const int nX = S_LEN * H_DIM;
  const int nE = NH_ * S_LEN * D_;
  cvt_f16_kernel<<<dim3((nX + 255) / 256), dim3(256), 0, stream>>>(x, xh, nX);
  cvt_f16_kernel<<<dim3((nE + 255) / 256), dim3(256), 0, stream>>>(E, Eh, nE);
  transpose_cvt_kernel<<<dim3(H3_ / 32, H_DIM / 32), dim3(32, 8), 0, stream>>>(
      W_attn, Wt_attn, H_DIM, H3_);
  transpose_cvt_kernel<<<dim3(H_DIM / 32, H_DIM / 32), dim3(32, 8), 0, stream>>>(
      W_proj, Wt_proj, H_DIM, H_DIM);

  qkv_gemm_kernel<<<dim3(H3_ / 64, S_LEN / 32), dim3(32), 0, stream>>>(
      xh, Wt_attn, b_attn, present, qh, kh, vt);

  flash_attn_rel_kernel<<<dim3(NH_, S_LEN / 16), dim3(32), 0, stream>>>(
      qh, kh, vt, Eh, aoh);

  proj_gemm_kernel<<<dim3(H_DIM / 64, S_LEN / 32), dim3(32), 0, stream>>>(
      aoh, Wt_proj, b_proj, out);
}
